// Decoder_81260781241014
// MI455X (gfx1250) — matrix operand order repaired and checked by reading, not compiled, not run on hardware
//
#include <hip/hip_runtime.h>

// ---------------------------------------------------------------------------
// Persistent cooperative LSTM-attention decoder for gfx1250 (MI455X).
// One kernel launch; 64 blocks x 256 threads (= 512 wave32). Device-wide
// barriers between phases. All GEMMs run on v_wmma_f32_16x16x32_bf16 with
// bf16-resident activations (no per-iteration cvt in the inner loop).
// ---------------------------------------------------------------------------

#define DEV __device__ __forceinline__

typedef __attribute__((ext_vector_type(16))) __bf16 v16bf;
typedef __attribute__((ext_vector_type(8)))  __bf16 v8bf;
typedef __attribute__((ext_vector_type(8)))  float  v8f;

constexpr int Bc  = 64;     // batch
constexpr int Dc  = 512;    // model dim
constexpr int Sc  = 512;    // source length
constexpr int Tc  = 64;     // target steps
constexpr int G4D = 2048;   // 4*D (gates)
constexpr int NBLK = 64;
constexpr int NTHR = 256;
constexpr int NTHREADS = NBLK * NTHR;   // 16384

struct Params {
  const int*   tokens;   // [T][B]
  const float* mem;      // [S][B][D]
  const int*   lens;     // [B]
  const float* emb;      // [V][D]
  const float* Wih0; const float* Whh0; const float* bih0; const float* bhh0;
  const float* Wih1; const float* Whh1; const float* bih1; const float* bhh1;
  const float* Wout;
  float* out_dec;        // [T][B][D]
  float* out_attn;       // [T][B][S]
  __bf16* wih0bf;        // [4D][2D]
  __bf16* whh0bf;        // [4D][D]
  __bf16* wih1bf;        // [4D][D]
  __bf16* whh1bf;        // [4D][D]
  __bf16* woutbf;        // [D][2D]
  __bf16* X0bf;          // [B][2D]  concat(emb_t, feed), bf16
  __bf16* X2bf;          // [B][2D]  concat(ctx, h1), bf16
  __bf16* h0bf;          // [B][D]   (layer-0 hidden, GEMM-only -> bf16 only)
  __bf16* h1bf;          // [B][D]   bf16 mirror of h1
  __bf16* feedbf;        // [B][D]   input-feed, bf16
  float* g0;             // [B][4D]
  float* g1;             // [B][4D]
  float* c0; float* c1;  // [B][D] cell states, f32
  float* h1;             // [B][D] f32 (attention needs full precision)
  float* p;              // [B][S]
  unsigned* cnt;         // barrier counter
};

// ---- device-wide barrier (all NBLK blocks co-resident) --------------------
DEV void grid_barrier(unsigned* cnt, unsigned& gen) {
  __threadfence();            // release my stores to device scope
  __syncthreads();
  gen += NBLK;
  if (threadIdx.x == 0) {
    __hip_atomic_fetch_add(cnt, 1u, __ATOMIC_RELEASE, __HIP_MEMORY_SCOPE_AGENT);
    while (__hip_atomic_load(cnt, __ATOMIC_ACQUIRE, __HIP_MEMORY_SCOPE_AGENT) < gen) {
      __builtin_amdgcn_s_sleep(2);
    }
  }
  __syncthreads();
  __threadfence();            // acquire side
}

DEV float sigf(float x)   { return 1.0f / (1.0f + __expf(-x)); }
DEV float tanh_f(float x) { float e = __expf(2.0f * x); return 1.0f - 2.0f / (e + 1.0f); }

DEV void cvt_arr(__bf16* dst, const float* src, int n, int gthread) {
  for (int i = gthread; i < n; i += NTHREADS) dst[i] = (__bf16)src[i];
}

// ---- one 16x16 tile accumulation over K with bf16 WMMA --------------------
// A: bf16 activations, row-major [*, lda], rows mb..mb+15.
//    A layout (matches B): lanes 0-15 row M=lane, elems K+0..15;
//    lanes 16-31 same rows, elems K+16..31 (contig).
//    -> two aligned 16B loads per lane per k-step, no conversions.
// W: bf16 weights, row-major [*, ldw] -> B(k,n) = W[nb+n][k].
//    B layout used here: lane n (0-15) K=0..15, lane n+16 K=16..31 (contig).
DEV void wmma_seg(v8f& acc, const __bf16* __restrict__ A, int lda,
                  const __bf16* __restrict__ W, int ldw, int K,
                  int mb, int nb, int lane) {
  const int rowA = mb + (lane & 15);
  const int ka   = (lane < 16) ? 0 : 16;
  const int nB   = nb + (lane & 15);
  const int kb0  = (lane < 16) ? 0 : 16;
  const __bf16* ap = A + (size_t)rowA * lda + ka;
  const __bf16* wp = W + (size_t)nB * ldw + kb0;
  for (int k = 0; k < K; k += 32) {
    const v8bf a_lo = *(const v8bf*)(ap + k);        // K = kb+ka .. +7
    const v8bf a_hi = *(const v8bf*)(ap + k + 8);    // K = kb+ka+8 .. +15
    const v16bf afrag = __builtin_shufflevector(
        a_lo, a_hi, 0, 1, 2, 3, 4, 5, 6, 7, 8, 9, 10, 11, 12, 13, 14, 15);
    const v16bf bfrag = *(const v16bf*)(wp + k);     // 32B aligned
    acc = __builtin_amdgcn_wmma_f32_16x16x32_bf16(
        /*neg_a=*/false, afrag, /*neg_b=*/false, bfrag,
        /*c_mod=*/(short)0, acc, /*reuse_a=*/false, /*reuse_b=*/false);
  }
}

// C/D layout: VGPR r -> row mb + r + (lane<16 ? 0 : 8), col nb + (lane&15)
DEV void store_tile(float* out, int ldc, int mb, int nb, int lane, const v8f& acc) {
  const int col = nb + (lane & 15);
  const int rb  = mb + ((lane < 16) ? 0 : 8);
#pragma unroll
  for (int r = 0; r < 8; ++r) out[(size_t)(rb + r) * ldc + col] = acc[r];
}

__global__ void __launch_bounds__(NTHR)
decoder_persistent(Params P) {
  const int tid     = threadIdx.x;
  const int bid     = blockIdx.x;
  const int gthread = bid * NTHR + tid;
  const int gwave   = gthread >> 5;
  const int lane    = tid & 31;
  unsigned  gen     = 0;

  __shared__ __align__(16) float sm_v[Dc];
  __shared__ float sm_red[NTHR];

  // ---- prologue: weights -> bf16 in L2-resident scratch; zero state ------
  cvt_arr(P.wih0bf, P.Wih0, G4D * 2 * Dc, gthread);
  cvt_arr(P.whh0bf, P.Whh0, G4D * Dc,     gthread);
  cvt_arr(P.wih1bf, P.Wih1, G4D * Dc,     gthread);
  cvt_arr(P.whh1bf, P.Whh1, G4D * Dc,     gthread);
  cvt_arr(P.woutbf, P.Wout, Dc * 2 * Dc,  gthread);
  for (int i = gthread; i < Bc * Dc; i += NTHREADS) {
    P.c0[i] = 0.f; P.c1[i] = 0.f; P.h1[i] = 0.f;
    P.h0bf[i] = (__bf16)0.f; P.h1bf[i] = (__bf16)0.f; P.feedbf[i] = (__bf16)0.f;
  }
  grid_barrier(P.cnt, gen);

  for (int t = 0; t < Tc; ++t) {
    // ---- Phase A: X0 = [emb_t | feed]  (bf16) ---------------------------
    for (int i = gthread; i < Bc * 2 * Dc; i += NTHREADS) {
      const int b = i >> 10;
      const int k = i & (2 * Dc - 1);
      __bf16 v;
      if (k < Dc) {
        const int tok = P.tokens[t * Bc + b];
        v = (__bf16)P.emb[(size_t)tok * Dc + k];
      } else {
        v = P.feedbf[b * Dc + (k - Dc)];
      }
      P.X0bf[i] = v;
    }
    grid_barrier(P.cnt, gen);

    // ---- Phase B: g0 = X0 @ Wih0^T + h0 @ Whh0^T  (4 x 128 tiles) -------
    {
      const int mb = (gwave & 3) << 4;
      const int nb = (gwave >> 2) << 4;
      v8f acc = {};
      wmma_seg(acc, P.X0bf, 2 * Dc, P.wih0bf, 2 * Dc, 2 * Dc, mb, nb, lane);
      wmma_seg(acc, P.h0bf, Dc,     P.whh0bf, Dc,     Dc,     mb, nb, lane);
      store_tile(P.g0, G4D, mb, nb, lane, acc);
    }
    grid_barrier(P.cnt, gen);

    // ---- Phase C: LSTM cell 0 (biases folded in here) -------------------
    for (int i = gthread; i < Bc * Dc; i += NTHREADS) {
      const int b = i >> 9;
      const int d = i & (Dc - 1);
      const float* g = P.g0 + (size_t)b * G4D;
      const float ig = sigf(  g[d]          + P.bih0[d]          + P.bhh0[d]);
      const float fg = sigf(  g[Dc + d]     + P.bih0[Dc + d]     + P.bhh0[Dc + d]);
      const float gg = tanh_f(g[2 * Dc + d] + P.bih0[2 * Dc + d] + P.bhh0[2 * Dc + d]);
      const float og = sigf(  g[3 * Dc + d] + P.bih0[3 * Dc + d] + P.bhh0[3 * Dc + d]);
      const float c  = fg * P.c0[i] + ig * gg;
      P.c0[i]   = c;
      P.h0bf[i] = (__bf16)(og * tanh_f(c));   // h0 consumed only by GEMMs
    }
    grid_barrier(P.cnt, gen);

    // ---- Phase D: g1 = h0 @ Wih1^T + h1 @ Whh1^T ------------------------
    {
      const int mb = (gwave & 3) << 4;
      const int nb = (gwave >> 2) << 4;
      v8f acc = {};
      wmma_seg(acc, P.h0bf, Dc, P.wih1bf, Dc, Dc, mb, nb, lane);
      wmma_seg(acc, P.h1bf, Dc, P.whh1bf, Dc, Dc, mb, nb, lane);
      store_tile(P.g1, G4D, mb, nb, lane, acc);
    }
    grid_barrier(P.cnt, gen);

    // ---- Phase E: LSTM cell 1 -------------------------------------------
    for (int i = gthread; i < Bc * Dc; i += NTHREADS) {
      const int b = i >> 9;
      const int d = i & (Dc - 1);
      const float* g = P.g1 + (size_t)b * G4D;
      const float ig = sigf(  g[d]          + P.bih1[d]          + P.bhh1[d]);
      const float fg = sigf(  g[Dc + d]     + P.bih1[Dc + d]     + P.bhh1[Dc + d]);
      const float gg = tanh_f(g[2 * Dc + d] + P.bih1[2 * Dc + d] + P.bhh1[2 * Dc + d]);
      const float og = sigf(  g[3 * Dc + d] + P.bih1[3 * Dc + d] + P.bhh1[3 * Dc + d]);
      const float c  = fg * P.c1[i] + ig * gg;
      const float h  = og * tanh_f(c);
      P.c1[i]   = c;
      P.h1[i]   = h;              // f32 for attention dot
      P.h1bf[i] = (__bf16)h;      // bf16 for recurrent GEMM + X2
    }
    grid_barrier(P.cnt, gen);

    // ---- Phase F: scores + masked softmax (block b = bid) ---------------
    {
      const int b = bid;
      for (int i = tid; i < Dc; i += NTHR) sm_v[i] = P.h1[b * Dc + i];
      __syncthreads();
      const int s0 = tid, s1 = tid + NTHR;
      float a0 = 0.f, a1 = 0.f;
      const float4* hv = (const float4*)sm_v;
      const float4* m0 = (const float4*)(P.mem + (size_t)s0 * (Bc * Dc) + b * Dc);
      const float4* m1 = (const float4*)(P.mem + (size_t)s1 * (Bc * Dc) + b * Dc);
#pragma unroll 4
      for (int j = 0; j < Dc / 4; ++j) {
        const float4 h4 = hv[j];
        const float4 x0 = m0[j];
        const float4 x1 = m1[j];
        a0 = fmaf(h4.x, x0.x, fmaf(h4.y, x0.y, fmaf(h4.z, x0.z, fmaf(h4.w, x0.w, a0))));
        a1 = fmaf(h4.x, x1.x, fmaf(h4.y, x1.y, fmaf(h4.z, x1.z, fmaf(h4.w, x1.w, a1))));
      }
      const int len = P.lens[b];
      const float v0 = (s0 < len) ? a0 : -1e9f;
      const float v1 = (s1 < len) ? a1 : -1e9f;
      sm_red[tid] = fmaxf(v0, v1);
      __syncthreads();
      for (int off = NTHR / 2; off > 0; off >>= 1) {
        if (tid < off) sm_red[tid] = fmaxf(sm_red[tid], sm_red[tid + off]);
        __syncthreads();
      }
      const float mx = sm_red[0];
      __syncthreads();
      const float e0 = __expf(v0 - mx);
      const float e1 = __expf(v1 - mx);
      sm_red[tid] = e0 + e1;
      __syncthreads();
      for (int off = NTHR / 2; off > 0; off >>= 1) {
        if (tid < off) sm_red[tid] += sm_red[tid + off];
        __syncthreads();
      }
      const float inv = 1.0f / sm_red[0];
      const float p0 = e0 * inv, p1 = e1 * inv;
      P.p[b * Sc + s0] = p0;
      P.p[b * Sc + s1] = p1;
      float* oa = P.out_attn + ((size_t)t * Bc + b) * Sc;
      oa[s0] = p0;
      oa[s1] = p1;
    }
    grid_barrier(P.cnt, gen);

    // ---- Phase G: ctx = p @ mem ; X2 = [ctx | h1]  (bf16, block b) ------
    {
      const int b = bid;
      for (int i = tid; i < Sc; i += NTHR) sm_v[i] = P.p[b * Sc + i];
      __syncthreads();
      const int d0 = tid, d1 = tid + NTHR;
      float a0 = 0.f, a1 = 0.f;
      const float* mbase = P.mem + b * Dc;
      for (int s = 0; s < Sc; ++s) {
        const float ps = sm_v[s];
        const float* mr = mbase + (size_t)s * (Bc * Dc);
        a0 = fmaf(ps, mr[d0], a0);
        a1 = fmaf(ps, mr[d1], a1);
      }
      __bf16* x2 = P.X2bf + (size_t)b * 2 * Dc;
      x2[d0]      = (__bf16)a0;
      x2[d1]      = (__bf16)a1;
      x2[Dc + d0] = P.h1bf[b * Dc + d0];
      x2[Dc + d1] = P.h1bf[b * Dc + d1];
    }
    grid_barrier(P.cnt, gen);

    // ---- Phase H: attn_h = tanh(X2 @ Wout^T) -> feed + dec_outs[t] ------
    if (gwave < 128) {                      // 4 x 32 tiles; wave-uniform branch
      const int mb = (gwave & 3) << 4;
      const int nb = (gwave >> 2) << 4;
      v8f acc = {};
      wmma_seg(acc, P.X2bf, 2 * Dc, P.woutbf, 2 * Dc, 2 * Dc, mb, nb, lane);
      const int col = nb + (lane & 15);
      const int rb  = mb + ((lane < 16) ? 0 : 8);
      float* od = P.out_dec + (size_t)t * Bc * Dc;
#pragma unroll
      for (int r = 0; r < 8; ++r) {
        const float v = tanh_f(acc[r]);
        const int row = rb + r;
        P.feedbf[row * Dc + col] = (__bf16)v;   // identical to cvt at X0-build
        od[(size_t)row * Dc + col] = v;
      }
    }
    grid_barrier(P.cnt, gen);
  }
}

// ---------------------------------------------------------------------------
extern "C" void kernel_launch(void* const* d_in, const int* in_sizes, int n_in,
                              void* d_out, int out_size, void* d_ws, size_t ws_size,
                              hipStream_t stream) {
  (void)in_sizes; (void)n_in; (void)out_size; (void)ws_size;
  Params P;
  P.tokens = (const int*)  d_in[0];
  P.mem    = (const float*)d_in[1];
  P.lens   = (const int*)  d_in[2];
  P.emb    = (const float*)d_in[3];
  P.Wih0   = (const float*)d_in[4];
  P.Whh0   = (const float*)d_in[5];
  P.bih0   = (const float*)d_in[6];
  P.bhh0   = (const float*)d_in[7];
  P.Wih1   = (const float*)d_in[8];
  P.Whh1   = (const float*)d_in[9];
  P.bih1   = (const float*)d_in[10];
  P.bhh1   = (const float*)d_in[11];
  P.Wout   = (const float*)d_in[12];

  float* out = (float*)d_out;
  P.out_dec  = out;
  P.out_attn = out + (size_t)Tc * Bc * Dc;

  char*  w   = (char*)d_ws;
  size_t off = 0;
  auto alloc = [&](size_t bytes) -> char* {
    off = (off + 255) & ~(size_t)255;
    char* p = w + off;
    off += bytes;
    return p;
  };
  P.wih0bf = (__bf16*)alloc((size_t)G4D * 2 * Dc * 2);
  P.whh0bf = (__bf16*)alloc((size_t)G4D * Dc * 2);
  P.wih1bf = (__bf16*)alloc((size_t)G4D * Dc * 2);
  P.whh1bf = (__bf16*)alloc((size_t)G4D * Dc * 2);
  P.woutbf = (__bf16*)alloc((size_t)Dc * 2 * Dc * 2);
  P.X0bf   = (__bf16*)alloc((size_t)Bc * 2 * Dc * 2);
  P.X2bf   = (__bf16*)alloc((size_t)Bc * 2 * Dc * 2);
  P.h0bf   = (__bf16*)alloc((size_t)Bc * Dc * 2);
  P.h1bf   = (__bf16*)alloc((size_t)Bc * Dc * 2);
  P.feedbf = (__bf16*)alloc((size_t)Bc * Dc * 2);
  P.g0   = (float*)alloc((size_t)Bc * G4D * 4);
  P.g1   = (float*)alloc((size_t)Bc * G4D * 4);
  P.c0   = (float*)alloc((size_t)Bc * Dc * 4);
  P.c1   = (float*)alloc((size_t)Bc * Dc * 4);
  P.h1   = (float*)alloc((size_t)Bc * Dc * 4);
  P.p    = (float*)alloc((size_t)Bc * Sc * 4);
  P.cnt  = (unsigned*)alloc(256);

  hipMemsetAsync(P.cnt, 0, sizeof(unsigned), stream);  // barrier counter = 0
  decoder_persistent<<<dim3(NBLK), dim3(NTHR), 0, stream>>>(P);
}
